// SingleHeadAttention_6133213298767
// MI455X (gfx1250) — matrix operand order repaired and checked by reading, not compiled, not run on hardware
//
#include <hip/hip_runtime.h>

typedef unsigned short u16;
typedef __bf16 bf16_t;
typedef bf16_t v16bf __attribute__((ext_vector_type(16)));
typedef float  v8f   __attribute__((ext_vector_type(8)));
typedef unsigned int u32x4 __attribute__((ext_vector_type(4)));
typedef int i32x4 __attribute__((ext_vector_type(4)));
typedef int i32x8 __attribute__((ext_vector_type(8)));

#if __has_builtin(__builtin_amdgcn_tensor_load_to_lds)
#define ATHENA_HAS_TDM 1
#else
#define ATHENA_HAS_TDM 0
#endif

// ---- fp32 -> bf16, round-to-nearest-even ----
__device__ __forceinline__ u16 f2bf(float f) {
  unsigned u = __builtin_bit_cast(unsigned, f);
  u += 0x7FFFu + ((u >> 16) & 1u);
  return (u16)(u >> 16);
}

// A-fragment (16x32 bf16): lane half h holds K = {8h..8h+7} U {8h+16..8h+23}.
// p points at &row[8h]; chunks at +0 and +16 elements (16B-aligned).
__device__ __forceinline__ v16bf frag_a(const u16* p) {
  union { v16bf v; uint4 q[2]; } u;
  u.q[0] = *(const uint4*)(p);
  u.q[1] = *(const uint4*)(p + 16);
  return u.v;
}
// B-fragment (32x16 bf16): lane half h holds K = {8h..8h+7} U {8h+16..8h+23}.
__device__ __forceinline__ v16bf frag_b(const u16* p) {
  union { v16bf v; uint4 q[2]; } u;
  u.q[0] = *(const uint4*)(p);
  u.q[1] = *(const uint4*)(p + 16);
  return u.v;
}

// ---------------------------------------------------------------------------
// TDM: DMA one 128-row x 32-col bf16 tile (row stride `ld` elements) from
// global memory into LDS, inserting 16B of padding after every 64B row so the
// LDS image has row pitch LDK=40 bf16 (matches the manual-staging layout).
// D# per CDNA5 ISA ch. 8: group0 = {count,lds_addr,global_addr,type=2},
// group1 = {data_size=2B, pad 16dw interval / 4dw amount, dims, stride}.
// Issued by a single wave; EXEC is ignored by TENSOR_LOAD_TO_LDS.
// ---------------------------------------------------------------------------
__device__ __forceinline__ void tdm_load_tile_128x32(const u16* gsrc, u16* ldst,
                                                     long long ld) {
#if ATHENA_HAS_TDM
  const unsigned long long ga = (unsigned long long)gsrc;
  const unsigned lds_off =
      (unsigned)(unsigned long long)(__attribute__((address_space(3))) u16*)ldst;
  u32x4 g0;
  g0[0] = 1u;                                   // count=1, no gather
  g0[1] = lds_off;                              // lds_addr (bytes)
  g0[2] = (unsigned)(ga & 0xFFFFFFFFu);         // global_addr[31:0]
  g0[3] = (unsigned)((ga >> 32) & 0x1FFFFFFu)   // global_addr[56:32]
          | (2u << 30);                         // type=2 ("image")
  i32x8 g1;
  g1[0] = (int)((1u << 16)      // data_size = 2 bytes
                | (1u << 20)    // pad_enable
                | (3u << 22)    // pad_interval: 16 dwords (64B = one row)
                | (3u << 25));  // pad_amount:   4 dwords (16B -> pitch 80B)
  g1[1] = (int)(32u << 16);     // tensor_dim0[15:0] = 32 (elements)
  g1[2] = (int)(128u << 16);    // tensor_dim0[31:16]=0 | tensor_dim1[15:0]=128
  g1[3] = (int)(32u << 16);     // tensor_dim1[31:16]=0 | tile_dim0 = 32
  g1[4] = (int)128u;            // tile_dim1 = 128 | tile_dim2 = 0
  g1[5] = (int)(unsigned)(ld & 0xFFFFFFFFll);   // tensor_dim0_stride[31:0]
  g1[6] = (int)(unsigned)((ld >> 32) & 0xFFFFll);  // stride[47:32] | d1stride lo = 0
  g1[7] = 0;
  const i32x4 z4 = {0, 0, 0, 0};
#if __clang_major__ >= 23
  const i32x8 z8 = {0, 0, 0, 0, 0, 0, 0, 0};
  __builtin_amdgcn_tensor_load_to_lds(g0, g1, z4, z4, z8, 0);
#else
  __builtin_amdgcn_tensor_load_to_lds(g0, g1, z4, z4, 0);
#endif
#else
  (void)gsrc; (void)ldst; (void)ld;
#endif
}

__device__ __forceinline__ void wait_tensorcnt0() {
#if __has_builtin(__builtin_amdgcn_s_wait_tensorcnt)
  __builtin_amdgcn_s_wait_tensorcnt(0);
#endif
}

// ---------------------------------------------------------------------------
// Generic WMMA bf16 GEMM:  C = scale * (A @ B) + bias
//   Block tile 128x128, BK=32, 8 waves (4x2), wave tile 32x64 (2x4 WMMA tiles)
//   A_F32 / B_F32 : global operand is f32 (converted to bf16 while staging)
//   B_NK          : B is stored [N][K] row-major (i.e. computes A @ B^T)
//   OUT_F32       : store f32, else bf16
// bf16 operands whose global layout matches the LDS layout are staged by the
// Tensor Data Mover (wave 0 -> A tile, wave 1 -> B tile) when available.
// Requires: M%128==0, N%128==0, K%32==0 (true for all uses here).
// ---------------------------------------------------------------------------
template <bool A_F32, bool B_F32, bool B_NK, bool OUT_F32>
__global__ __launch_bounds__(256) void gemm_wmma(
    const void* __restrict__ Av, long long lda, long long strideA,
    const void* __restrict__ Bv, long long ldb, long long strideB,
    void* __restrict__ Cv, long long ldc, long long strideC,
    const float* __restrict__ bias, int K, float scale) {
  constexpr int BM = 128, BN = 128, BK = 32, LDK = 40;  // LDK pads banks
  constexpr bool useTdmA = ATHENA_HAS_TDM && !A_F32;           // A: [m][k] bf16
  constexpr bool useTdmB = ATHENA_HAS_TDM && !B_F32 && B_NK;   // B: [n][k] bf16
  __shared__ u16 sA[BM * LDK];
  __shared__ u16 sB[BN * LDK];

  const int tid  = threadIdx.x;
  const int lane = tid & 31;
  const int wave = tid >> 5;
  const int wm   = wave >> 1;   // 0..3  (32 rows each)
  const int wn   = wave & 1;    // 0..1  (64 cols each)
  const int half = lane >> 4;   // lane half
  const int l15  = lane & 15;

  const long long m0 = (long long)blockIdx.y * BM;
  const long long n0 = (long long)blockIdx.x * BN;
  const long long z  = blockIdx.z;

  const float* Af = A_F32 ? (const float*)Av + z * strideA : (const float*)nullptr;
  const u16*   Ah = A_F32 ? (const u16*)nullptr : (const u16*)Av + z * strideA;
  const float* Bf = B_F32 ? (const float*)Bv + z * strideB : (const float*)nullptr;
  const u16*   Bh = B_F32 ? (const u16*)nullptr : (const u16*)Bv + z * strideB;

  const v8f vzero = {0.f, 0.f, 0.f, 0.f, 0.f, 0.f, 0.f, 0.f};
  v8f acc[2][4];
#pragma unroll
  for (int i = 0; i < 2; ++i)
#pragma unroll
    for (int j = 0; j < 4; ++j) acc[i][j] = vzero;

  for (int k0 = 0; k0 < K; k0 += BK) {
    // ---- stage A tile (BM x BK) as bf16 into LDS [m][k] ----
    if constexpr (useTdmA) {
      if (wave == 0) tdm_load_tile_128x32(Ah + m0 * lda + k0, sA, lda);
    } else {
      const int c  = (tid & 7) * 4;  // k within tile
      const int r0 = tid >> 3;       // row within tile
#pragma unroll
      for (int rr = 0; rr < 4; ++rr) {
        const int row = r0 + rr * 32;
        u16* dst = &sA[row * LDK + c];
        if constexpr (A_F32) {
          const float4 f = *(const float4*)(Af + (m0 + row) * lda + k0 + c);
          ushort4 h;
          h.x = f2bf(f.x); h.y = f2bf(f.y); h.z = f2bf(f.z); h.w = f2bf(f.w);
          *(ushort4*)dst = h;
        } else {
          *(ushort4*)dst = *(const ushort4*)(Ah + (m0 + row) * lda + k0 + c);
        }
      }
    }
    // ---- stage B tile as bf16 into LDS [n][k] ----
    if constexpr (useTdmB) {
      if (wave == 1) tdm_load_tile_128x32(Bh + n0 * ldb + k0, sB, ldb);
    } else if constexpr (B_NK) {
      const int c  = (tid & 7) * 4;
      const int r0 = tid >> 3;
#pragma unroll
      for (int rr = 0; rr < 4; ++rr) {
        const int row = r0 + rr * 32;
        u16* dst = &sB[row * LDK + c];
        if constexpr (B_F32) {
          const float4 f = *(const float4*)(Bf + (n0 + row) * ldb + k0 + c);
          ushort4 h;
          h.x = f2bf(f.x); h.y = f2bf(f.y); h.z = f2bf(f.z); h.w = f2bf(f.w);
          *(ushort4*)dst = h;
        } else {
          *(ushort4*)dst = *(const ushort4*)(Bh + (n0 + row) * ldb + k0 + c);
        }
      }
    } else {
      const int nc  = (tid & 31) * 4;  // n within tile
      const int kr0 = tid >> 5;        // k within tile
#pragma unroll
      for (int rr = 0; rr < 4; ++rr) {
        const int kk = kr0 + rr * 8;
        if constexpr (B_F32) {
          const float4 f = *(const float4*)(Bf + (long long)(k0 + kk) * ldb + n0 + nc);
          sB[(nc + 0) * LDK + kk] = f2bf(f.x);
          sB[(nc + 1) * LDK + kk] = f2bf(f.y);
          sB[(nc + 2) * LDK + kk] = f2bf(f.z);
          sB[(nc + 3) * LDK + kk] = f2bf(f.w);
        } else {
          const ushort4 h = *(const ushort4*)(Bh + (long long)(k0 + kk) * ldb + n0 + nc);
          sB[(nc + 0) * LDK + kk] = h.x;
          sB[(nc + 1) * LDK + kk] = h.y;
          sB[(nc + 2) * LDK + kk] = h.z;
          sB[(nc + 3) * LDK + kk] = h.w;
        }
      }
    }
    // TDM-issuing waves drain TENSORcnt before the barrier so every wave
    // observes the DMA'd tiles after s_barrier.
    if constexpr (useTdmA) {
      if (wave == 0) wait_tensorcnt0();
    }
    if constexpr (useTdmB) {
      if (wave == 1) wait_tensorcnt0();
    }
    __syncthreads();

    // ---- fragments + 8 WMMAs per wave ----
    v16bf af[2], bg[4];
#pragma unroll
    for (int i = 0; i < 2; ++i)
      af[i] = frag_a(&sA[(wm * 32 + i * 16 + l15) * LDK + half * 8]);
#pragma unroll
    for (int j = 0; j < 4; ++j)
      bg[j] = frag_b(&sB[(wn * 64 + j * 16 + l15) * LDK + half * 8]);

#pragma unroll
    for (int i = 0; i < 2; ++i)
#pragma unroll
      for (int j = 0; j < 4; ++j)
        acc[i][j] = __builtin_amdgcn_wmma_f32_16x16x32_bf16(
            false, af[i], false, bg[j], (short)0, acc[i][j], false, false);

    __syncthreads();
  }

  // ---- epilogue: C/D layout: n = lane&15, m = 8*half + vgpr ----
  float* Cf = OUT_F32 ? (float*)Cv + z * strideC : (float*)nullptr;
  u16*   Ch = OUT_F32 ? (u16*)nullptr : (u16*)Cv + z * strideC;
#pragma unroll
  for (int j = 0; j < 4; ++j) {
    const long long n = n0 + wn * 64 + j * 16 + l15;
    const float bv = bias ? bias[n] : 0.0f;
#pragma unroll
    for (int i = 0; i < 2; ++i) {
      const long long mb = m0 + wm * 32 + i * 16 + half * 8;
#pragma unroll
      for (int r = 0; r < 8; ++r) {
        const float v = acc[i][j][r] * scale + bv;
        if constexpr (OUT_F32) Cf[(mb + r) * ldc + n] = v;
        else                   Ch[(mb + r) * ldc + n] = f2bf(v);
      }
    }
  }
}

// ---------------------------------------------------------------------------
// Row softmax: one 256-thread block per row of 4096 f32 scores -> bf16 probs
// ---------------------------------------------------------------------------
__global__ __launch_bounds__(256) void softmax_rows(
    const float* __restrict__ S, u16* __restrict__ P, int ncols) {
  const long long row = blockIdx.x;
  const float* sr = S + row * (long long)ncols;
  u16* pr = P + row * (long long)ncols;
  const int tid = threadIdx.x, lane = tid & 31, wave = tid >> 5;

  float v[16];
#pragma unroll
  for (int i = 0; i < 16; ++i) v[i] = sr[tid + (i << 8)];

  float mx = v[0];
#pragma unroll
  for (int i = 1; i < 16; ++i) mx = fmaxf(mx, v[i]);
#pragma unroll
  for (int off = 16; off >= 1; off >>= 1) mx = fmaxf(mx, __shfl_xor(mx, off, 32));

  __shared__ float red[8];
  __shared__ float bval;
  if (lane == 0) red[wave] = mx;
  __syncthreads();
  if (tid == 0) {
    float m = red[0];
#pragma unroll
    for (int w = 1; w < 8; ++w) m = fmaxf(m, red[w]);
    bval = m;
  }
  __syncthreads();
  mx = bval;

  float s = 0.0f;
#pragma unroll
  for (int i = 0; i < 16; ++i) {
    v[i] = __expf(v[i] - mx);
    s += v[i];
  }
#pragma unroll
  for (int off = 16; off >= 1; off >>= 1) s += __shfl_xor(s, off, 32);
  if (lane == 0) red[wave] = s;
  __syncthreads();
  if (tid == 0) {
    float t = 0.0f;
#pragma unroll
    for (int w = 0; w < 8; ++w) t += red[w];
    bval = 1.0f / t;
  }
  __syncthreads();
  const float inv = bval;
#pragma unroll
  for (int i = 0; i < 16; ++i) pr[tid + (i << 8)] = f2bf(v[i] * inv);
}

// ---------------------------------------------------------------------------
extern "C" void kernel_launch(void* const* d_in, const int* in_sizes, int n_in,
                              void* d_out, int out_size, void* d_ws, size_t ws_size,
                              hipStream_t stream) {
  (void)in_sizes; (void)n_in; (void)out_size; (void)ws_size;
  const float* x    = (const float*)d_in[0];  // (4,4096,4096)
  const float* Wqkv = (const float*)d_in[1];  // (4096,3072)
  const float* bqkv = (const float*)d_in[2];  // (3072,)
  const float* Wo   = (const float*)d_in[3];  // (1024,4096)
  const float* bo   = (const float*)d_in[4];  // (4096,)

  const long long Bn = 4, S = 4096, H = 4096, Dh = 1024, F = 3072;
  const long long M1 = Bn * S;  // 16384

  // workspace carve-out
  char* ws = (char*)d_ws;
  auto al = [](size_t v) { return (v + 255) & ~(size_t)255; };
  size_t off = 0;
  u16* qkvb = (u16*)(ws + off);       off = al(off + (size_t)(Bn * S * F) * sizeof(u16));   // 100 MB
  float* scores = (float*)(ws + off); off = al(off + (size_t)(Bn * S * S) * sizeof(float)); // 256 MB
  u16* probs = (u16*)(ws + off);      off = al(off + (size_t)(Bn * S * S) * sizeof(u16));   // 134 MB
  u16* ob = (u16*)(ws + off);         off = al(off + (size_t)(Bn * S * Dh) * sizeof(u16));  // 34 MB

  const dim3 blk(256);

  // 1) qkv = x @ Wqkv + bqkv   (f32 in, bf16 out; interleaved [b][s][3*1024])
  gemm_wmma<true, true, false, false>
      <<<dim3((unsigned)(F / 128), (unsigned)(M1 / 128), 1), blk, 0, stream>>>(
          x, H, 0, Wqkv, F, 0, qkvb, F, 0, bqkv, (int)H, 1.0f);

  // 2) scores = (1/sqrt(d)) * q @ k^T   per batch (B as [N][K] = k rows; A+B via TDM)
  gemm_wmma<false, false, true, true>
      <<<dim3((unsigned)(S / 128), (unsigned)(S / 128), (unsigned)Bn), blk, 0, stream>>>(
          qkvb, F, S * F, qkvb + Dh, F, S * F, scores, S, S * S, nullptr, (int)Dh, 0.03125f);

  // 3) softmax over keys, f32 -> bf16 probs
  softmax_rows<<<dim3((unsigned)(Bn * S)), blk, 0, stream>>>(scores, probs, (int)S);

  // 4) O = P @ V   per batch (A via TDM)
  gemm_wmma<false, false, false, false>
      <<<dim3((unsigned)(Dh / 128), (unsigned)(S / 128), (unsigned)Bn), blk, 0, stream>>>(
          probs, S, S * S, qkvb + 2 * Dh, F, S * F, ob, Dh, S * Dh, nullptr, (int)S, 1.0f);

  // 5) out = O @ Wo + bo   (bf16 A via TDM, f32 B, f32 out)
  gemm_wmma<false, true, false, true>
      <<<dim3((unsigned)(H / 128), (unsigned)(M1 / 128), 1), blk, 0, stream>>>(
          ob, Dh, 0, Wo, H, 0, d_out, H, 0, bo, (int)Dh, 1.0f);
}
